// MLPMoE_16217796510266
// MI455X (gfx1250) — compile-verified, matrix operand order checked by reading, not run on hardware
//
#include <hip/hip_runtime.h>
#include <hip/hip_bf16.h>
#include <math.h>

#define T_TOK 2048
#define H_DIM 768
#define I_DIM 2048
#define E_NUM 8

typedef __attribute__((ext_vector_type(16))) __bf16 v16bf;
typedef __attribute__((ext_vector_type(8)))  __bf16 v8bf;
typedef __attribute__((ext_vector_type(2)))  __bf16 v2bf;
typedef __attribute__((ext_vector_type(8)))  float  v8f;
typedef __attribute__((ext_vector_type(4)))  float  v4f;
typedef __attribute__((ext_vector_type(2)))  float  v2f;

union FragU { v16bf v; v8bf h[2]; };

// Load one 16x32 bf16 WMMA operand fragment from an LDS tile stored
// [16 rows][stride elems]. Per ISA 7.12.2 (16-bit A 16x32):
//   lane<16 : row=lane,    K = {0..7, 16..23}
//   lane>=16: row=lane-16, K = {8..15, 24..31}
// B comes from a column-major ([n][k]) LDS tile with the same pattern.
__device__ __forceinline__ v16bf lds_frag(const __bf16* base, int stride, int lane) {
  int r  = lane & 15;
  int kb = (lane >> 4) << 3;
  const __bf16* p = base + r * stride + kb;
  FragU f;
  f.h[0] = *(const v8bf*)(p);
  f.h[1] = *(const v8bf*)(p + 16);
  return f.v;
}

__device__ __forceinline__ float sigmoidf_(float x) { return 1.0f / (1.0f + __expf(-x)); }

#define WMMA_BF16(A, B, C) \
  __builtin_amdgcn_wmma_f32_16x16x32_bf16(false, (A), false, (B), (short)0, (C), false, false)

// ---------------------------------------------------------------- init
__global__ void moe_init_kernel(int* counts) {
  if (threadIdx.x < E_NUM) counts[threadIdx.x] = 0;
}

// ---------------------------------------------------------------- router : one wave per token
__global__ void moe_router_kernel(const float* __restrict__ x,
                                  const float* __restrict__ rw,
                                  int* __restrict__ idx,
                                  float* __restrict__ score,
                                  int* __restrict__ counts,
                                  int* __restrict__ list) {
  int wave = threadIdx.x >> 5;
  int lane = threadIdx.x & 31;
  int t = blockIdx.x * 4 + wave;
  float acc[E_NUM];
#pragma unroll
  for (int e = 0; e < E_NUM; ++e) acc[e] = 0.0f;
  for (int h = lane; h < H_DIM; h += 32) {
    float xv = x[t * H_DIM + h];
    const float* wr = rw + h * E_NUM;
#pragma unroll
    for (int e = 0; e < E_NUM; ++e) acc[e] += xv * wr[e];
  }
#pragma unroll
  for (int e = 0; e < E_NUM; ++e) {
#pragma unroll
    for (int off = 16; off > 0; off >>= 1)
      acc[e] += __shfl_xor(acc[e], off, 32);
  }
  if (lane == 0) {
    int best = 0; float bv = acc[0];
#pragma unroll
    for (int e = 1; e < E_NUM; ++e) { if (acc[e] > bv) { bv = acc[e]; best = e; } }
    idx[t] = best;
    score[t] = sigmoidf_(bv);
    int pos = atomicAdd(&counts[best], 1);
    list[best * T_TOK + pos] = t;
  }
}

// ---------------------------------------------------------------- finalize (deterministic reduction)
__global__ void moe_finalize_kernel(const int* __restrict__ idx,
                                    const float* __restrict__ score,
                                    const int* __restrict__ counts,
                                    int* __restrict__ offsets,
                                    float* __restrict__ loss_out) {
  __shared__ float red[256 * E_NUM];
  int tid = threadIdx.x;
  float s[E_NUM];
#pragma unroll
  for (int e = 0; e < E_NUM; ++e) s[e] = 0.0f;
  for (int t = tid; t < T_TOK; t += 256) s[idx[t]] += score[t];
#pragma unroll
  for (int e = 0; e < E_NUM; ++e) red[tid * E_NUM + e] = s[e];
  __syncthreads();
  for (int str = 128; str > 0; str >>= 1) {
    if (tid < str) {
#pragma unroll
      for (int e = 0; e < E_NUM; ++e)
        red[tid * E_NUM + e] += red[(tid + str) * E_NUM + e];
    }
    __syncthreads();
  }
  if (tid == 0) {
    int off = 0; float loss = 0.0f;
#pragma unroll
    for (int e = 0; e < E_NUM; ++e) {
      offsets[e] = off;
      off += counts[e];
      loss += (float)counts[e] * red[e];
    }
    offsets[E_NUM] = off;
    loss_out[0] = loss * (0.001f * (float)E_NUM) / ((float)T_TOK * (float)T_TOK);
  }
}

// ================================================================ gate/up GEMMs
// Block tile M=128 x N=32 (I cols), fused gate & up.
// 8 waves arranged 4M x 2N; each wave owns 32 rows x 16 cols for BOTH matrices:
// per K-step: 4 A ds_load_b128 + 4 B ds_load_b128 feed 4 WMMAs (2 ds/wmma).
struct GateUpAcc { v8f cg0, cg1, cu0, cu1; };

__device__ __forceinline__ void gateup_step(const __bf16* As, const __bf16* Bg,
                                            const __bf16* Bu, int wm, int wn, int lane,
                                            GateUpAcc& a) {
  v16bf a0 = lds_frag(As + wm * 40, 40, lane);
  v16bf a1 = lds_frag(As + (wm + 16) * 40, 40, lane);
  v16bf gf = lds_frag(Bg + wn * 40, 40, lane);
  v16bf uf = lds_frag(Bu + wn * 40, 40, lane);
  a.cg0 = WMMA_BF16(a0, gf, a.cg0);
  a.cg1 = WMMA_BF16(a1, gf, a.cg1);
  a.cu0 = WMMA_BF16(a0, uf, a.cu0);
  a.cu1 = WMMA_BF16(a1, uf, a.cu1);
}

// B staging: 32k x 32n per matrix; thread packs 2 k's per v2bf (b32 LDS stores).
__device__ __forceinline__ void stage_b32x32(const float* __restrict__ w, int ldw,
                                             __bf16* Bs, int k0_tid, int n_tid) {
  const float* p0 = w + (size_t)k0_tid * ldw + n_tid;
  v2f a = *(const v2f*)p0;
  v2f b = *(const v2f*)(p0 + ldw);
  v2bf q0 = { (__bf16)a[0], (__bf16)b[0] };
  v2bf q1 = { (__bf16)a[1], (__bf16)b[1] };
  *(v2bf*)(Bs + n_tid * 40 + k0_tid) = q0;
  *(v2bf*)(Bs + (n_tid + 1) * 40 + k0_tid) = q1;
}

__global__ void moe_gateup_shared_kernel(const float* __restrict__ x,
                                         const float* __restrict__ wg,
                                         const float* __restrict__ wu,
                                         __bf16* __restrict__ hs) {
  __shared__ __attribute__((aligned(16))) __bf16 As[128 * 40];
  __shared__ __attribute__((aligned(16))) __bf16 Bg[32 * 40];
  __shared__ __attribute__((aligned(16))) __bf16 Bu[32 * 40];
  int tid = threadIdx.x;
  int row0 = blockIdx.x * 128;
  int n0 = blockIdx.y * 32;
  int wave = tid >> 5, lane = tid & 31;
  int wm = (wave >> 1) * 32, wn = (wave & 1) * 16;
  GateUpAcc acc; acc.cg0 = {}; acc.cg1 = {}; acc.cu0 = {}; acc.cu1 = {};
  int ar = tid >> 1, ak = (tid & 1) * 16;       // A: 16 floats/thread
  int bk = (tid >> 4) * 2, bn = (tid & 15) * 2; // B: 2x2 cell/thread/matrix

  for (int k0 = 0; k0 < H_DIM; k0 += 32) {
    const float* xp = x + (size_t)(row0 + ar) * H_DIM + k0 + ak;
    __bf16* ap = As + ar * 40 + ak;
#pragma unroll
    for (int c = 0; c < 2; ++c) {
      v4f a0 = *(const v4f*)(xp + c * 8);
      v4f a1 = *(const v4f*)(xp + c * 8 + 4);
#pragma unroll
      for (int j = 0; j < 4; ++j) { ap[c * 8 + j] = (__bf16)a0[j]; ap[c * 8 + 4 + j] = (__bf16)a1[j]; }
    }
    stage_b32x32(wg + (size_t)k0 * I_DIM + n0, I_DIM, Bg, bk, bn);
    stage_b32x32(wu + (size_t)k0 * I_DIM + n0, I_DIM, Bu, bk, bn);
    if (k0 + 32 < H_DIM) {
      __builtin_prefetch(wg + (size_t)(k0 + 32 + bk) * I_DIM + n0 + bn, 0, 0);
      __builtin_prefetch(wu + (size_t)(k0 + 32 + bk) * I_DIM + n0 + bn, 0, 0);
    }
    __syncthreads();
    gateup_step(As, Bg, Bu, wm, wn, lane, acc);
    __syncthreads();
  }
  int nn = n0 + wn + (lane & 15);
  int mbase = wm + ((lane >> 4) << 3);
#pragma unroll
  for (int r = 0; r < 8; ++r) {
    float g0 = acc.cg0[r], u0 = acc.cu0[r];
    float g1 = acc.cg1[r], u1 = acc.cu1[r];
    hs[(size_t)(row0 + mbase + r) * I_DIM + nn]      = (__bf16)(g0 * sigmoidf_(g0) * u0);
    hs[(size_t)(row0 + 16 + mbase + r) * I_DIM + nn] = (__bf16)(g1 * sigmoidf_(g1) * u1);
  }
}

__global__ void moe_gateup_routed_kernel(const float* __restrict__ x,
                                         const float* __restrict__ score,
                                         const int* __restrict__ list,
                                         const int* __restrict__ counts,
                                         const int* __restrict__ offsets,
                                         const float* __restrict__ gate_w,
                                         const float* __restrict__ up_w,
                                         __bf16* __restrict__ hr) {
  int e = blockIdx.x >> 4;            // T/128 = 16 M-tiles per expert
  int mt = blockIdx.x & 15;
  int cnt = counts[e];
  int rowstart = mt * 128;
  if (rowstart >= cnt) return;        // nearly all of the 8x grid exits here
  int offBase = offsets[e];
  const float* wg = gate_w + (size_t)e * H_DIM * I_DIM;
  const float* wu = up_w + (size_t)e * H_DIM * I_DIM;

  __shared__ __attribute__((aligned(16))) __bf16 As[128 * 40];
  __shared__ __attribute__((aligned(16))) __bf16 Bg[32 * 40];
  __shared__ __attribute__((aligned(16))) __bf16 Bu[32 * 40];
  int tid = threadIdx.x;
  int n0 = blockIdx.y * 32;
  int wave = tid >> 5, lane = tid & 31;
  int wm = (wave >> 1) * 32, wn = (wave & 1) * 16;
  GateUpAcc acc; acc.cg0 = {}; acc.cg1 = {}; acc.cu0 = {}; acc.cu1 = {};
  int ar = tid >> 1, ak = (tid & 1) * 16;
  int bk = (tid >> 4) * 2, bn = (tid & 15) * 2;

  int slot = rowstart + ar;
  bool valid = slot < cnt;
  int tok = 0; float sc = 0.0f;
  if (valid) { tok = list[e * T_TOK + slot]; sc = score[tok]; }

  for (int k0 = 0; k0 < H_DIM; k0 += 32) {
    __bf16* ap = As + ar * 40 + ak;
    if (valid) {
      const float* xp = x + (size_t)tok * H_DIM + k0 + ak;
#pragma unroll
      for (int c = 0; c < 2; ++c) {
        v4f a0 = *(const v4f*)(xp + c * 8);
        v4f a1 = *(const v4f*)(xp + c * 8 + 4);
#pragma unroll
        for (int j = 0; j < 4; ++j) {
          ap[c * 8 + j] = (__bf16)(a0[j] * sc);
          ap[c * 8 + 4 + j] = (__bf16)(a1[j] * sc);
        }
      }
    } else {
      v8bf z = {};
      *(v8bf*)(ap) = z;
      *(v8bf*)(ap + 8) = z;
    }
    stage_b32x32(wg + (size_t)k0 * I_DIM + n0, I_DIM, Bg, bk, bn);
    stage_b32x32(wu + (size_t)k0 * I_DIM + n0, I_DIM, Bu, bk, bn);
    if (k0 + 32 < H_DIM) {
      __builtin_prefetch(wg + (size_t)(k0 + 32 + bk) * I_DIM + n0 + bn, 0, 0);
      __builtin_prefetch(wu + (size_t)(k0 + 32 + bk) * I_DIM + n0 + bn, 0, 0);
    }
    __syncthreads();
    gateup_step(As, Bg, Bu, wm, wn, lane, acc);
    __syncthreads();
  }
  int nn = n0 + wn + (lane & 15);
  int mbase = wm + ((lane >> 4) << 3);
#pragma unroll
  for (int r = 0; r < 8; ++r) {
    int so0 = rowstart + mbase + r;
    int so1 = so0 + 16;
    if (so0 < cnt) {
      float g = acc.cg0[r], u = acc.cu0[r];
      hr[(size_t)(offBase + so0) * I_DIM + nn] = (__bf16)(g * sigmoidf_(g) * u);
    }
    if (so1 < cnt) {
      float g = acc.cg1[r], u = acc.cu1[r];
      hr[(size_t)(offBase + so1) * I_DIM + nn] = (__bf16)(g * sigmoidf_(g) * u);
    }
  }
}

// ================================================================ down GEMMs
// Block tile M=128 x N=64; 8 waves arranged 4M x 2N; each wave owns 32x32:
// per K-step: 4 A + 4 B ds_load_b128 feed 4 WMMAs (2 ds/wmma).
struct DownAcc { v8f c00, c01, c10, c11; };

__device__ __forceinline__ void down_step(const __bf16* As, const __bf16* Bs,
                                          int wm, int wn, int lane, DownAcc& a) {
  v16bf a0 = lds_frag(As + wm * 40, 40, lane);
  v16bf a1 = lds_frag(As + (wm + 16) * 40, 40, lane);
  v16bf b0 = lds_frag(Bs + wn * 40, 40, lane);
  v16bf b1 = lds_frag(Bs + (wn + 16) * 40, 40, lane);
  a.c00 = WMMA_BF16(a0, b0, a.c00);
  a.c01 = WMMA_BF16(a0, b1, a.c01);
  a.c10 = WMMA_BF16(a1, b0, a.c10);
  a.c11 = WMMA_BF16(a1, b1, a.c11);
}

// B staging: 32k x 64n; thread covers 2 k's x 4 n's, packed b32 stores.
__device__ __forceinline__ void stage_b32x64(const float* __restrict__ w, int ldw,
                                             __bf16* Bs, int k0_tid, int n_tid) {
  const float* p0 = w + (size_t)k0_tid * ldw + n_tid;
  v4f a = *(const v4f*)p0;
  v4f b = *(const v4f*)(p0 + ldw);
#pragma unroll
  for (int j = 0; j < 4; ++j) {
    v2bf q = { (__bf16)a[j], (__bf16)b[j] };
    *(v2bf*)(Bs + (n_tid + j) * 40 + k0_tid) = q;
  }
}

__global__ void moe_down_shared_kernel(const __bf16* __restrict__ hs,
                                       const float* __restrict__ wd,
                                       float* __restrict__ out) {
  __shared__ __attribute__((aligned(16))) __bf16 As[128 * 40];
  __shared__ __attribute__((aligned(16))) __bf16 Bs[64 * 40];
  int tid = threadIdx.x;
  int row0 = blockIdx.x * 128;
  int n0 = blockIdx.y * 64;
  int wave = tid >> 5, lane = tid & 31;
  int wm = (wave >> 1) * 32, wn = (wave & 1) * 32;
  DownAcc acc; acc.c00 = {}; acc.c01 = {}; acc.c10 = {}; acc.c11 = {};
  int ar = tid >> 1, ak = (tid & 1) * 16;       // A: 16 bf16/thread
  int bk = (tid >> 4) * 2, bn = (tid & 15) * 4; // B: 2k x 4n /thread

  for (int k0 = 0; k0 < I_DIM; k0 += 32) {
    const __bf16* hp = hs + (size_t)(row0 + ar) * I_DIM + k0 + ak;
    *(v8bf*)(As + ar * 40 + ak) = *(const v8bf*)(hp);
    *(v8bf*)(As + ar * 40 + ak + 8) = *(const v8bf*)(hp + 8);
    stage_b32x64(wd + (size_t)k0 * H_DIM + n0, H_DIM, Bs, bk, bn);
    if (k0 + 32 < I_DIM)
      __builtin_prefetch(wd + (size_t)(k0 + 32 + bk) * H_DIM + n0 + bn, 0, 0);
    __syncthreads();
    down_step(As, Bs, wm, wn, lane, acc);
    __syncthreads();
  }
  int nn0 = n0 + wn + (lane & 15);
  int mbase = wm + ((lane >> 4) << 3);
#pragma unroll
  for (int r = 0; r < 8; ++r) {
    out[(size_t)(row0 + mbase + r) * H_DIM + nn0]           = acc.c00[r];
    out[(size_t)(row0 + mbase + r) * H_DIM + nn0 + 16]      = acc.c01[r];
    out[(size_t)(row0 + 16 + mbase + r) * H_DIM + nn0]      = acc.c10[r];
    out[(size_t)(row0 + 16 + mbase + r) * H_DIM + nn0 + 16] = acc.c11[r];
  }
}

__global__ void moe_down_routed_kernel(const __bf16* __restrict__ hr,
                                       const float* __restrict__ down_w,
                                       const int* __restrict__ list,
                                       const int* __restrict__ counts,
                                       const int* __restrict__ offsets,
                                       float* __restrict__ out) {
  int e = blockIdx.x >> 4;
  int mt = blockIdx.x & 15;
  int cnt = counts[e];
  int rowstart = mt * 128;
  if (rowstart >= cnt) return;
  int offBase = offsets[e];
  const float* wd = down_w + (size_t)e * I_DIM * H_DIM;

  __shared__ __attribute__((aligned(16))) __bf16 As[128 * 40];
  __shared__ __attribute__((aligned(16))) __bf16 Bs[64 * 40];
  int tid = threadIdx.x;
  int n0 = blockIdx.y * 64;
  int wave = tid >> 5, lane = tid & 31;
  int wm = (wave >> 1) * 32, wn = (wave & 1) * 32;
  DownAcc acc; acc.c00 = {}; acc.c01 = {}; acc.c10 = {}; acc.c11 = {};
  int ar = tid >> 1, ak = (tid & 1) * 16;
  int bk = (tid >> 4) * 2, bn = (tid & 15) * 4;
  int slot = rowstart + ar;
  bool valid = slot < cnt;

  for (int k0 = 0; k0 < I_DIM; k0 += 32) {
    __bf16* ap = As + ar * 40 + ak;
    if (valid) {
      const __bf16* hp = hr + (size_t)(offBase + slot) * I_DIM + k0 + ak;
      *(v8bf*)(ap) = *(const v8bf*)(hp);
      *(v8bf*)(ap + 8) = *(const v8bf*)(hp + 8);
    } else {
      v8bf z = {};
      *(v8bf*)(ap) = z;
      *(v8bf*)(ap + 8) = z;
    }
    stage_b32x64(wd + (size_t)k0 * H_DIM + n0, H_DIM, Bs, bk, bn);
    if (k0 + 32 < I_DIM)
      __builtin_prefetch(wd + (size_t)(k0 + 32 + bk) * H_DIM + n0 + bn, 0, 0);
    __syncthreads();
    down_step(As, Bs, wm, wn, lane, acc);
    __syncthreads();
  }
  int nn0 = n0 + wn + (lane & 15);
  int mbase = wm + ((lane >> 4) << 3);
#pragma unroll
  for (int r = 0; r < 8; ++r) {
    int so0 = rowstart + mbase + r;
    int so1 = so0 + 16;
    if (so0 < cnt) {
      int tok = list[e * T_TOK + so0];
      out[(size_t)tok * H_DIM + nn0]      += acc.c00[r];   // unique writer
      out[(size_t)tok * H_DIM + nn0 + 16] += acc.c01[r];
    }
    if (so1 < cnt) {
      int tok = list[e * T_TOK + so1];
      out[(size_t)tok * H_DIM + nn0]      += acc.c10[r];
      out[(size_t)tok * H_DIM + nn0 + 16] += acc.c11[r];
    }
  }
}

// ---------------------------------------------------------------- launch
extern "C" void kernel_launch(void* const* d_in, const int* in_sizes, int n_in,
                              void* d_out, int out_size, void* d_ws, size_t ws_size,
                              hipStream_t stream) {
  (void)in_sizes; (void)n_in; (void)out_size; (void)ws_size;
  const float* x        = (const float*)d_in[0];
  const float* router_w = (const float*)d_in[1];
  const float* gate_w   = (const float*)d_in[2];
  const float* up_w     = (const float*)d_in[3];
  const float* down_w   = (const float*)d_in[4];
  const float* sh_gate  = (const float*)d_in[5];
  const float* sh_up    = (const float*)d_in[6];
  const float* sh_down  = (const float*)d_in[7];
  float* out = (float*)d_out;

  char* ws = (char*)d_ws;
  int*    counts  = (int*)(ws + 0);                 // 8 ints
  int*    offsets = (int*)(ws + 64);                // 9 ints
  int*    idx     = (int*)(ws + 128);               // T ints
  float*  score   = (float*)(ws + 128 + 4 * T_TOK); // T floats
  int*    list    = (int*)(ws + 128 + 8 * T_TOK);   // E*T ints
  size_t  base    = 128 + 8 * (size_t)T_TOK + 4 * (size_t)E_NUM * T_TOK;
  base = (base + 255) & ~(size_t)255;
  __bf16* hs = (__bf16*)(ws + base);                                 // T*I bf16
  __bf16* hr = (__bf16*)(ws + base + 2 * (size_t)T_TOK * I_DIM);     // T*I bf16 (compacted)

  moe_init_kernel<<<1, 32, 0, stream>>>(counts);
  moe_router_kernel<<<T_TOK / 4, 128, 0, stream>>>(x, router_w, idx, score, counts, list);
  moe_finalize_kernel<<<1, 256, 0, stream>>>(idx, score, counts, offsets, out + (size_t)T_TOK * H_DIM);
  moe_gateup_shared_kernel<<<dim3(T_TOK / 128, I_DIM / 32), 256, 0, stream>>>(x, sh_gate, sh_up, hs);
  moe_gateup_routed_kernel<<<dim3(E_NUM * (T_TOK / 128), I_DIM / 32), 256, 0, stream>>>(
      x, score, list, counts, offsets, gate_w, up_w, hr);
  moe_down_shared_kernel<<<dim3(T_TOK / 128, H_DIM / 64), 256, 0, stream>>>(hs, sh_down, out);
  moe_down_routed_kernel<<<dim3(E_NUM * (T_TOK / 128), H_DIM / 64), 256, 0, stream>>>(
      hr, down_w, list, counts, offsets, out);
}
